// DeepSeekSparseAttention_81277961109551
// MI455X (gfx1250) — matrix operand order repaired and checked by reading, not compiled, not run on hardware
//
#include <hip/hip_runtime.h>
#include <hip/hip_bf16.h>

// ---------------------------------------------------------------------------
// DeepSeek sparse attention for gfx1250 (MI455X).
// S=768, E=512, H=8, DK=64, TOPK=384, B=1. Output f32 (768*512).
// ---------------------------------------------------------------------------

#define S_LEN 768
#define EMB   512
#define NH    8
#define DK    64
#define TOPK  384

typedef __bf16 bf16_t;
typedef __attribute__((ext_vector_type(16))) bf16_t v16bf;
typedef __attribute__((ext_vector_type(8)))  float  v8f;
typedef __attribute__((ext_vector_type(4)))  unsigned int u32x4;
typedef __attribute__((ext_vector_type(8)))  int    i32x8;
typedef __attribute__((ext_vector_type(4)))  int    i32x4;

union FragBF {
    v16bf          bf;
    unsigned int   u32[8];
    unsigned short us[16];
};

static __device__ __forceinline__ unsigned short f32_to_bf16_rne(float f) {
    union { float f; unsigned u; } v; v.f = f;
    unsigned u = v.u;
    unsigned r = u + 0x7FFFu + ((u >> 16) & 1u);
    return (unsigned short)(r >> 16);
}

// Load a 32-byte fragment (8 dwords of packed bf16 pairs) from any pointer.
static __device__ __forceinline__ void load_frag32(FragBF& f, const unsigned short* p) {
    *(uint4*)&f.u32[0] = *(const uint4*)p;
    *(uint4*)&f.u32[4] = *(const uint4*)(p + 8);
}

#if __has_builtin(__builtin_amdgcn_tensor_load_to_lds)
#define HAVE_TDM 1
#else
#define HAVE_TDM 0
#endif

#if HAVE_TDM
// TDM: load a 2D tile of (64 rows x 32 bf16) from a (Np x K) bf16 tensor into
// LDS at lds_off. D# packed per CDNA5 ISA.
static __device__ __forceinline__ void tdm_load_b_panel(const unsigned short* gsrc,
                                                        unsigned int lds_off,
                                                        int K, int Np) {
    unsigned long long ga = (unsigned long long)gsrc;
    u32x4 g0;
    g0[0] = 1u;                                   // count=1, user descriptor
    g0[1] = lds_off;                              // lds_addr
    g0[2] = (unsigned)ga;                         // global_addr[31:0]
    g0[3] = (unsigned)((ga >> 32) & 0x01FFFFFFu)  // global_addr[56:32]
          | 0x80000000u;                          // type=2 ("image")
    i32x8 g1;
    g1[0] = 0x00010000;                           // data_size = 2 bytes
    g1[1] = (int)((K & 0xFFFF) << 16);            // tensor_dim0[15:0] @ bits63:48
    g1[2] = (int)(((K >> 16) & 0xFFFF) | ((Np & 0xFFFF) << 16)); // dim0 hi | dim1 lo
    g1[3] = (int)(((Np >> 16) & 0xFFFF) | (32 << 16));           // dim1 hi | tile_dim0=32
    g1[4] = 64;                                   // tile_dim1 = 64 rows (tile_dim2=0)
    g1[5] = K;                                    // tensor_dim0_stride (lo32)
    g1[6] = 0;                                    // stride hi / dim1_stride lo (unused 2D)
    g1[7] = 0;
    i32x4 g2 = {0, 0, 0, 0};
    i32x4 g3 = {0, 0, 0, 0};
#if __clang_major__ >= 23
    i32x8 g4 = {0, 0, 0, 0, 0, 0, 0, 0};
    __builtin_amdgcn_tensor_load_to_lds(g0, g1, g2, g3, g4, 0);
#else
    __builtin_amdgcn_tensor_load_to_lds(g0, g1, g2, g3, 0);
#endif
}

static __device__ __forceinline__ void tdm_wait() {
#if __has_builtin(__builtin_amdgcn_s_wait_tensorcnt)
    __builtin_amdgcn_s_wait_tensorcnt(0);
#else
    asm volatile("s_wait_tensorcnt 0x0" ::: "memory");
#endif
}
#endif // HAVE_TDM

// ---------------------------------------------------------------------------
// Elementwise f32 -> bf16
// ---------------------------------------------------------------------------
__global__ void k_cvt_bf16(const float* __restrict__ in,
                           unsigned short* __restrict__ out, int n) {
    int i = blockIdx.x * blockDim.x + threadIdx.x;
    if (i < n) out[i] = f32_to_bf16_rne(in[i]);
}

// W is (K x N) row-major f32 -> Wt (Np x K) row-major bf16 (pad rows untouched;
// they only feed discarded output columns).
__global__ void k_transpose_bf16(const float* __restrict__ W,
                                 unsigned short* __restrict__ Wt,
                                 int K, int N) {
    int i = blockIdx.x * blockDim.x + threadIdx.x;
    if (i < K * N) {
        int k = i / N, n = i % N;
        Wt[n * K + k] = f32_to_bf16_rne(W[i]);
    }
}

// ---------------------------------------------------------------------------
// bf16 WMMA GEMM: C[M,N] = A[M,K](bf16,rm) * Bt[Np,K](bf16,rm)^T + bias
// Workgroup (8 waves) covers a 128-row x 64-col block. Each wave: one 16-row
// x 64-col strip = 4 accumulators, 4 WMMAs per K-step. The 64x32 B panel per
// K-step is staged into LDS once per workgroup via the Tensor Data Mover.
// Requires: M % 128 == 0, Np % 64 == 0, K % 32 == 0.
// ---------------------------------------------------------------------------
__global__ __launch_bounds__(256)
void k_gemm_bf16_wmma(const unsigned short* __restrict__ A,
                      const unsigned short* __restrict__ Bt,
                      const float* __restrict__ bias,
                      float* __restrict__ Cf,
                      unsigned short* __restrict__ Cbf,
                      int M, int N, int Np, int K) {
    __shared__ unsigned short Bs[64 * 32];   // 4 KB B panel

    const int stripsX  = Np >> 6;
    const int blockRow = blockIdx.x / stripsX;
    const int strip    = blockIdx.x % stripsX;
    const int col0     = strip << 6;
    const int waveId   = threadIdx.x >> 5;
    const int lane     = threadIdx.x & 31;
    const int g        = lane >> 4;
    const int row0     = blockRow * 128 + waveId * 16;
    const int mA       = row0 + (lane & 15);

#if !HAVE_TDM
    const int tid      = threadIdx.x;
    const int stage_n  = tid >> 2;   // 0..63 panel row
    const int stage_c  = tid & 3;    // 16B chunk within row
#endif

    v8f acc0 = {}, acc1 = {}, acc2 = {}, acc3 = {};

    for (int k0 = 0; k0 < K; k0 += 32) {
#if HAVE_TDM
        if (waveId == 0) {
            tdm_load_b_panel(Bt + (size_t)col0 * K + k0,
                             (unsigned int)(unsigned long long)&Bs[0], K, Np);
            tdm_wait();
        }
#else
        *(uint4*)&Bs[stage_n * 32 + stage_c * 8] =
            *(const uint4*)(Bt + (size_t)(col0 + stage_n) * K + k0 + stage_c * 8);
#endif
        __syncthreads();

        // A fragment (global, two contiguous 16B chunks):
        // u32[v] holds k = 8g + 2v (v<4) and k = 16 + 8g + 2(v-4) (v>=4)
        FragBF a;
        const unsigned short* arow = A + (size_t)mA * K + k0 + 8 * g;
        if (k0 + 64 < K) __builtin_prefetch(arow + 64, 0, 1);
        load_frag32(a, arow);   // wait: chunk2 must come from arow+16, see below

        // fix second chunk: k = 16+8g.. lives at ushort offset 16 from arow
        *(uint4*)&a.u32[4] = *(const uint4*)(arow + 16);

        // B fragments from LDS: panel row = local col, 16B chunks at k=8g and k=16+8g
#pragma unroll
        for (int c = 0; c < 4; ++c) {
            FragBF b;
            const unsigned short* bl = &Bs[(c * 16 + (lane & 15)) * 32 + g * 8];
            *(uint4*)&b.u32[0] = *(const uint4*)(bl);
            *(uint4*)&b.u32[4] = *(const uint4*)(bl + 16);
            switch (c) {
            case 0: acc0 = __builtin_amdgcn_wmma_f32_16x16x32_bf16(
                        false, a.bf, false, b.bf, (short)0, acc0, false, false); break;
            case 1: acc1 = __builtin_amdgcn_wmma_f32_16x16x32_bf16(
                        false, a.bf, false, b.bf, (short)0, acc1, false, false); break;
            case 2: acc2 = __builtin_amdgcn_wmma_f32_16x16x32_bf16(
                        false, a.bf, false, b.bf, (short)0, acc2, false, false); break;
            case 3: acc3 = __builtin_amdgcn_wmma_f32_16x16x32_bf16(
                        false, a.bf, false, b.bf, (short)0, acc3, false, false); break;
            }
        }
        __syncthreads();   // all waves done reading Bs before next overwrite
    }

    // store 4 tiles
#pragma unroll
    for (int c = 0; c < 4; ++c) {
        const v8f* accp = (c == 0) ? &acc0 : (c == 1) ? &acc1 : (c == 2) ? &acc2 : &acc3;
        int nB = col0 + c * 16 + (lane & 15);
        if (nB < N) {
            float bv = bias ? bias[nB] : 0.0f;
#pragma unroll
            for (int e = 0; e < 8; ++e) {
                int m = row0 + e + 8 * g;
                float r = (*accp)[e] + bv;
                if (Cf)  Cf[(size_t)m * N + nB]  = r;
                if (Cbf) Cbf[(size_t)m * N + nB] = f32_to_bf16_rne(r);
            }
        }
    }
}

// ---------------------------------------------------------------------------
// Indexer score: iscore[s,t] = sum_h relu(q_idx[h,s,:] . k_idx[t,:]) * w_idx[s,h]
// ---------------------------------------------------------------------------
__global__ __launch_bounds__(256)
void k_indexer_score_wmma(const unsigned short* __restrict__ Qi,
                          const unsigned short* __restrict__ Ki,
                          const float* __restrict__ Wid,
                          float* __restrict__ iscore) {
    const int tilesX = S_LEN >> 4;     // 48
    const int wid = blockIdx.x * (blockDim.x >> 5) + (threadIdx.x >> 5);
    if (wid >= tilesX * tilesX) return;

    const int ty = wid / tilesX, tx = wid % tilesX;
    const int s0 = ty << 4, t0 = tx << 4;
    const int lane = threadIdx.x & 31;
    const int g    = lane >> 4;
    const int mrow = s0 + (lane & 15);
    const int ncol = t0 + (lane & 15);

    float fin[8] = {0, 0, 0, 0, 0, 0, 0, 0};

    for (int h = 0; h < NH; ++h) {
        v8f acc = {};
#pragma unroll
        for (int step = 0; step < 2; ++step) {
            const int k0 = step * 32;
            FragBF a, b;
            const unsigned short* arow = Qi + (size_t)mrow * EMB + h * DK + k0 + 8 * g;
            *(uint4*)&a.u32[0] = *(const uint4*)(arow);
            *(uint4*)&a.u32[4] = *(const uint4*)(arow + 16);
            const unsigned short* brow = Ki + (size_t)ncol * DK + k0 + 8 * g;
            *(uint4*)&b.u32[0] = *(const uint4*)(brow);
            *(uint4*)&b.u32[4] = *(const uint4*)(brow + 16);
            acc = __builtin_amdgcn_wmma_f32_16x16x32_bf16(
                false, a.bf, false, b.bf, (short)0, acc, false, false);
        }
#pragma unroll
        for (int e = 0; e < 8; ++e) {
            float w = Wid[(s0 + e + 8 * g) * NH + h];
            fin[e] = fmaf(fmaxf(acc[e], 0.0f), w, fin[e]);
        }
    }
#pragma unroll
    for (int e = 0; e < 8; ++e)
        iscore[(size_t)(s0 + e + 8 * g) * S_LEN + ncol] = fin[e];
}

// ---------------------------------------------------------------------------
// Top-K (set-equivalent): bitonic sort 1024 (val,idx) pairs in LDS per row.
// ---------------------------------------------------------------------------
__global__ __launch_bounds__(512)
void k_topk_bitonic(const float* __restrict__ iscore,
                    int* __restrict__ topk) {
    __shared__ float vals[1024];
    __shared__ int   idxs[1024];
    const int s   = blockIdx.x;
    const int tid = threadIdx.x;

    for (int i = tid; i < 1024; i += 512) {
        vals[i] = (i < S_LEN) ? iscore[(size_t)s * S_LEN + i] : -__builtin_inff();
        idxs[i] = i;
    }
    for (unsigned ks = 2; ks <= 1024; ks <<= 1) {
        for (unsigned j = ks >> 1; j > 0; j >>= 1) {
            __syncthreads();
            for (unsigned i = tid; i < 1024; i += 512) {
                unsigned ixj = i ^ j;
                if (ixj > i) {
                    bool desc = ((i & ks) == 0);
                    float a = vals[i], b = vals[ixj];
                    bool sw = desc ? (a < b) : (a > b);
                    if (sw) {
                        vals[i] = b; vals[ixj] = a;
                        int t = idxs[i]; idxs[i] = idxs[ixj]; idxs[ixj] = t;
                    }
                }
            }
        }
    }
    __syncthreads();
    for (int j = tid; j < TOPK; j += 512)
        topk[(size_t)s * TOPK + j] = idxs[j];
}

// ---------------------------------------------------------------------------
// Sparse attention for one (s, h).
// ---------------------------------------------------------------------------
__global__ __launch_bounds__(256)
void k_sparse_attn(const float* __restrict__ Q,
                   const float* __restrict__ K,
                   const float* __restrict__ V,
                   const int* __restrict__ topk,
                   unsigned short* __restrict__ attn_bf) {
    __shared__ float qs[DK];
    __shared__ float logits[TOPK];
    __shared__ float red[256];
    __shared__ float pacc[4][DK];
    __shared__ float s_max, s_inv;

    const int s   = blockIdx.x >> 3;
    const int h   = blockIdx.x & 7;
    const int tid = threadIdx.x;
    const float scaling = 0.125f;   // 1/sqrt(64)

    if (tid < DK) qs[tid] = Q[(size_t)s * EMB + h * DK + tid];
    __syncthreads();

    float lmax = -__builtin_inff();
    for (int j = tid; j < TOPK; j += 256) {
        int t = topk[(size_t)s * TOPK + j];
        const float* kp = K + (size_t)t * EMB + h * DK;
        float d = 0.0f;
#pragma unroll
        for (int dd = 0; dd < DK; ++dd) d = fmaf(qs[dd], kp[dd], d);
        d *= scaling;
        logits[j] = d;
        lmax = fmaxf(lmax, d);
    }
    red[tid] = lmax; __syncthreads();
    for (int off = 128; off > 0; off >>= 1) {
        if (tid < off) red[tid] = fmaxf(red[tid], red[tid + off]);
        __syncthreads();
    }
    if (tid == 0) s_max = red[0];
    __syncthreads();

    float lsum = 0.0f;
    float mv = s_max;
    for (int j = tid; j < TOPK; j += 256) {
        float e = __expf(logits[j] - mv);
        logits[j] = e;
        lsum += e;
    }
    red[tid] = lsum; __syncthreads();
    for (int off = 128; off > 0; off >>= 1) {
        if (tid < off) red[tid] += red[tid + off];
        __syncthreads();
    }
    if (tid == 0) s_inv = 1.0f / red[0];
    __syncthreads();

    const int d = tid & 63;
    const int c = tid >> 6;
    float a = 0.0f;
    for (int j = c; j < TOPK; j += 4) {
        int t = topk[(size_t)s * TOPK + j];
        a = fmaf(logits[j], V[(size_t)t * EMB + h * DK + d], a);
    }
    pacc[c][d] = a;
    __syncthreads();
    if (tid < DK) {
        float r = (pacc[0][tid] + pacc[1][tid] + pacc[2][tid] + pacc[3][tid]) * s_inv;
        attn_bf[(size_t)s * EMB + h * DK + tid] = f32_to_bf16_rne(r);
    }
}

// ---------------------------------------------------------------------------
// Host launch
// ---------------------------------------------------------------------------
extern "C" void kernel_launch(void* const* d_in, const int* in_sizes, int n_in,
                              void* d_out, int out_size, void* d_ws, size_t ws_size,
                              hipStream_t stream) {
    const float* x    = (const float*)d_in[0];
    const float* Wq   = (const float*)d_in[1];
    const float* bq   = (const float*)d_in[2];
    const float* Wk   = (const float*)d_in[3];
    const float* bk   = (const float*)d_in[4];
    const float* Wv   = (const float*)d_in[5];
    const float* bv   = (const float*)d_in[6];
    const float* Wo   = (const float*)d_in[7];
    const float* bo   = (const float*)d_in[8];
    const float* iqW  = (const float*)d_in[9];
    const float* iqb  = (const float*)d_in[10];
    const float* ikW  = (const float*)d_in[11];
    const float* ikb  = (const float*)d_in[12];
    const float* wpW  = (const float*)d_in[13];
    const float* wpb  = (const float*)d_in[14];

    size_t off = 0;
    auto alloc = [&](size_t bytes) -> void* {
        void* p = (char*)d_ws + off;
        off += (bytes + 255) & ~(size_t)255;
        return p;
    };

    unsigned short* xb    = (unsigned short*)alloc((size_t)S_LEN * EMB * 2);
    unsigned short* WqT   = (unsigned short*)alloc((size_t)EMB * EMB * 2);
    unsigned short* WkT   = (unsigned short*)alloc((size_t)EMB * EMB * 2);
    unsigned short* WvT   = (unsigned short*)alloc((size_t)EMB * EMB * 2);
    unsigned short* WoT   = (unsigned short*)alloc((size_t)EMB * EMB * 2);
    unsigned short* iqWT  = (unsigned short*)alloc((size_t)EMB * EMB * 2);
    unsigned short* ikWT  = (unsigned short*)alloc((size_t)64  * EMB * 2);   // Np=64
    unsigned short* wpWT  = (unsigned short*)alloc((size_t)64  * EMB * 2);   // Np=64 (N=8 padded)
    float*          Qf    = (float*)alloc((size_t)S_LEN * EMB * 4);
    float*          Kf    = (float*)alloc((size_t)S_LEN * EMB * 4);
    float*          Vf    = (float*)alloc((size_t)S_LEN * EMB * 4);
    unsigned short* qidxB = (unsigned short*)alloc((size_t)S_LEN * EMB * 2);
    unsigned short* kidxB = (unsigned short*)alloc((size_t)S_LEN * DK * 2);
    float*          widxF = (float*)alloc((size_t)S_LEN * NH * 4);
    float*          isc   = (float*)alloc((size_t)S_LEN * S_LEN * 4);
    int*            topk  = (int*)alloc((size_t)S_LEN * TOPK * 4);
    unsigned short* attnB = (unsigned short*)alloc((size_t)S_LEN * EMB * 2);

    // 1) x -> bf16
    {
        int n = S_LEN * EMB;
        k_cvt_bf16<<<(n + 255) / 256, 256, 0, stream>>>(x, xb, n);
    }
    // 2) transpose+convert all weights to (Np x K) bf16
    {
        int n = EMB * EMB;
        k_transpose_bf16<<<(n + 255) / 256, 256, 0, stream>>>(Wq,  WqT,  EMB, EMB);
        k_transpose_bf16<<<(n + 255) / 256, 256, 0, stream>>>(Wk,  WkT,  EMB, EMB);
        k_transpose_bf16<<<(n + 255) / 256, 256, 0, stream>>>(Wv,  WvT,  EMB, EMB);
        k_transpose_bf16<<<(n + 255) / 256, 256, 0, stream>>>(Wo,  WoT,  EMB, EMB);
        k_transpose_bf16<<<(n + 255) / 256, 256, 0, stream>>>(iqW, iqWT, EMB, EMB);
        int n2 = EMB * DK;
        k_transpose_bf16<<<(n2 + 255) / 256, 256, 0, stream>>>(ikW, ikWT, EMB, DK);
        int n3 = EMB * NH;
        k_transpose_bf16<<<(n3 + 255) / 256, 256, 0, stream>>>(wpW, wpWT, EMB, NH);
    }

    auto gemm = [&](const unsigned short* A, const unsigned short* Bt,
                    const float* bias, float* Cf, unsigned short* Cbf,
                    int M, int N, int Np, int K) {
        int blocks = (M / 128) * (Np / 64);
        k_gemm_bf16_wmma<<<blocks, 256, 0, stream>>>(A, Bt, bias, Cf, Cbf, M, N, Np, K);
    };

    // 3) projections (WMMA + TDM-staged B panels)
    gemm(xb, WqT,  bq,  Qf,    nullptr, S_LEN, EMB, EMB, EMB);
    gemm(xb, WkT,  bk,  Kf,    nullptr, S_LEN, EMB, EMB, EMB);
    gemm(xb, WvT,  bv,  Vf,    nullptr, S_LEN, EMB, EMB, EMB);
    gemm(xb, iqWT, iqb, nullptr, qidxB, S_LEN, EMB, EMB, EMB);
    gemm(xb, ikWT, ikb, nullptr, kidxB, S_LEN, DK,  64,  EMB);
    gemm(xb, wpWT, wpb, widxF, nullptr, S_LEN, NH,  64,  EMB);

    // 4) indexer score (WMMA, fused relu + head weighting)
    {
        int waves  = (S_LEN / 16) * (S_LEN / 16);    // 2304
        int blocks = (waves + 7) / 8;
        k_indexer_score_wmma<<<blocks, 256, 0, stream>>>(qidxB, kidxB, widxF, isc);
    }

    // 5) top-k per query row
    k_topk_bitonic<<<S_LEN, 512, 0, stream>>>(isc, topk);

    // 6) sparse attention per (s, h)
    k_sparse_attn<<<S_LEN * NH, 256, 0, stream>>>(Qf, Kf, Vf, topk, attnB);

    // 7) output projection (WMMA) -> d_out f32
    gemm(attnB, WoT, bo, (float*)d_out, nullptr, S_LEN, EMB, EMB, EMB);
}
